// BermMatrixLayer_85504208928778
// MI455X (gfx1250) — matrix operand order repaired and checked by reading, not compiled, not run on hardware
//
#include <hip/hip_runtime.h>
#include <hip/hip_bf16.h>
#include <math.h>

#define B_    4
#define S_    2048
#define HID_  1024
#define H_    16
#define D_    16
#define HV_   64
#define BH_   (B_*H_)        // 64
#define NCOL_ (H_*D_*D_)     // 4096
#define M_    (B_*S_)        // 8192
#define CHK_  16             // scan chunk length
#define NCH_  (S_/CHK_)      // 128 chunks

typedef __attribute__((ext_vector_type(2)))  float   v2f;
typedef __attribute__((ext_vector_type(8)))  float   v8f;
typedef __attribute__((ext_vector_type(8)))  __bf16  v8bf;
typedef __attribute__((ext_vector_type(16))) __bf16  v16bf;
union V16BF { v16bf v; v8bf h[2]; };

__device__ __forceinline__ unsigned short f2bf(float f) {
  unsigned int u = __float_as_uint(f);
  u += 0x7FFFu + ((u >> 16) & 1u);       // round-to-nearest-even
  return (unsigned short)(u >> 16);
}

// ---------------- K0a: hidden_states f32 -> bf16 ----------------
__global__ void k_cvt_hs(const float* __restrict__ in, unsigned short* __restrict__ out, int n) {
  int i = blockIdx.x * blockDim.x + threadIdx.x;
  if (i < n) out[i] = f2bf(in[i]);
}

// ---------------- K0b: W_mat f32 [K][N] -> bf16 transposed [N][K] ----------------
__global__ __launch_bounds__(256) void k_cvt_wt(const float* __restrict__ W,
                                                unsigned short* __restrict__ Wt) {
  __shared__ float tile[32][33];
  const int n0 = blockIdx.x * 32, k0 = blockIdx.y * 32;
  const int tx = threadIdx.x, ty = threadIdx.y;   // 32 x 8
  for (int i = 0; i < 32; i += 8)
    tile[ty + i][tx] = W[(size_t)(k0 + ty + i) * NCOL_ + n0 + tx];
  __syncthreads();
  for (int i = 0; i < 32; i += 8)
    Wt[(size_t)(n0 + ty + i) * HID_ + k0 + tx] = f2bf(tile[tx][ty + i]);
}

// ---------------- K1: bf16 WMMA GEMM, output scattered to [bh][s][16][16] ----------------
#define TM 128
#define TN 128
#define TK 32
#define LDA 40   // padded LDS row pitch (ushorts) -> bank-conflict-reducing, keeps 16B align
__global__ __launch_bounds__(256) void k_gemm(
    const unsigned short* __restrict__ Ab,   // [M_][HID_] bf16
    const unsigned short* __restrict__ Btb,  // [NCOL_][HID_] bf16 (W transposed)
    const float* __restrict__ bias,          // [NCOL_]
    float* __restrict__ Mout)                // [BH_][S_][256] f32
{
  __shared__ __align__(16) unsigned short As[TM * LDA];
  __shared__ __align__(16) unsigned short Bs[TN * LDA];
  const int tid  = threadIdx.x;
  const int lane = tid & 31, w = tid >> 5;
  const int wm = w & 1, wn = w >> 1;              // waves: 2 along M, 4 along N
  const int m0 = blockIdx.y * TM, n0 = blockIdx.x * TN;
  const int hlf = lane >> 4, r16 = lane & 15;

  v8f acc[4][2];
  for (int i = 0; i < 4; ++i) for (int j = 0; j < 2; ++j) acc[i][j] = (v8f){};

  const int lr = tid >> 1;          // staging row 0..127
  const int lc = (tid & 1) * 16;    // staging k-half 0/16

  for (int k0 = 0; k0 < HID_; k0 += TK) {
    { // cooperative tile staging: 32B per thread per tile
      const uint4* sa = reinterpret_cast<const uint4*>(Ab  + (size_t)(m0 + lr) * HID_ + k0 + lc);
      uint4* da = reinterpret_cast<uint4*>(As + lr * LDA + lc);
      da[0] = sa[0]; da[1] = sa[1];
      const uint4* sb = reinterpret_cast<const uint4*>(Btb + (size_t)(n0 + lr) * HID_ + k0 + lc);
      uint4* db = reinterpret_cast<uint4*>(Bs + lr * LDA + lc);
      db[0] = sb[0]; db[1] = sb[1];
      if (k0 + TK < HID_) {   // global_prefetch next K tiles
        __builtin_prefetch(Ab  + (size_t)(m0 + lr) * HID_ + k0 + TK + lc, 0, 1);
        __builtin_prefetch(Btb + (size_t)(n0 + lr) * HID_ + k0 + TK + lc, 0, 1);
      }
    }
    __syncthreads();
    V16BF afr[4], bfr[2];
    for (int mt = 0; mt < 4; ++mt) {           // A lane layout: row=lane&15, K = 8h..8h+7, 16+8h..
      const unsigned short* p = As + (wm * 64 + mt * 16 + r16) * LDA;
      afr[mt].h[0] = *reinterpret_cast<const v8bf*>(p + 8 * hlf);
      afr[mt].h[1] = *reinterpret_cast<const v8bf*>(p + 16 + 8 * hlf);
    }
    for (int nt = 0; nt < 2; ++nt) {           // B lane layout: col=lane&15, K = 8h..8h+7, 16+8h..
      const unsigned short* p = Bs + (wn * 32 + nt * 16 + r16) * LDA + 8 * hlf;
      bfr[nt].h[0] = *reinterpret_cast<const v8bf*>(p);
      bfr[nt].h[1] = *reinterpret_cast<const v8bf*>(p + 16);
    }
    for (int mt = 0; mt < 4; ++mt)
      for (int nt = 0; nt < 2; ++nt)
        acc[mt][nt] = __builtin_amdgcn_wmma_f32_16x16x32_bf16(
            false, afr[mt].v, false, bfr[nt].v, (short)0, acc[mt][nt], false, false);
    __syncthreads();
  }
  // epilogue: +bias, scatter into [bh][s][i*16+j] layout
  for (int nt = 0; nt < 2; ++nt) {
    const int Ncol = n0 + wn * 32 + nt * 16 + r16;
    const float bb_ = bias[Ncol];
    const int hh = Ncol >> 8, ij = Ncol & 255;
    for (int mt = 0; mt < 4; ++mt)
      for (int r = 0; r < 8; ++r) {
        const int R = m0 + wm * 64 + mt * 16 + r + 8 * hlf;   // D layout: M=r (+8 upper half)
        const int bb = R >> 11, ss = R & 2047;
        Mout[((size_t)(bb * 16 + hh) * S_ + ss) * 256 + ij] = acc[mt][nt][r] + bb_;
      }
  }
}

// ---------------- K2: Frobenius-normalize each 16x16 block in place ----------------
__global__ __launch_bounds__(256) void k_norm(float* __restrict__ Mbuf) {
  const int lane = threadIdx.x & 31, w = threadIdx.x >> 5;
  const int blk = blockIdx.x * 8 + w;          // 0..131071
  float* p = Mbuf + (size_t)blk * 256 + lane * 8;
  float4 a = *reinterpret_cast<float4*>(p);
  float4 b = *reinterpret_cast<float4*>(p + 4);
  float s = a.x*a.x + a.y*a.y + a.z*a.z + a.w*a.w
          + b.x*b.x + b.y*b.y + b.z*b.z + b.w*b.w;
  for (int off = 16; off > 0; off >>= 1) s += __shfl_xor(s, off, 32);
  const float inv = 1.0f / sqrtf(s);
  a.x*=inv; a.y*=inv; a.z*=inv; a.w*=inv;
  b.x*=inv; b.y*=inv; b.z*=inv; b.w*=inv;
  *reinterpret_cast<float4*>(p)     = a;
  *reinterpret_cast<float4*>(p + 4) = b;
}

// ---------------- K3a: per-chunk transfer matrices via f32 WMMA ----------------
// A_i = a_i*M_i + (1-a_i)I (lr), or with M^T (rl, reversed order). T_c = A_{c*16+15}...A_{c*16}
__global__ __launch_bounds__(32) void k_chunkT(
    const float* __restrict__ Mn, const float* __restrict__ mask,
    float* __restrict__ Tg)          // [2][BH_][NCH_][256]
{
  const int c = blockIdx.x, bh = blockIdx.y, dir = blockIdx.z;
  const int lane = threadIdx.x;
  const int r16 = lane & 15, hlf = lane >> 4;
  const int b = bh >> 4;
  __shared__ float T[2][16][17];
  for (int e = lane; e < 256; e += 32) T[0][e >> 4][e & 15] = ((e >> 4) == (e & 15)) ? 1.f : 0.f;
  __syncthreads();
  int cur = 0;
  for (int k = 0; k < CHK_; ++k) {
    const int i = c * CHK_ + k;
    const int s = dir ? (S_ - 1 - i) : i;
    const float aS = mask[b * S_ + s], om = 1.f - aS;
    const float* Ms = Mn + ((size_t)bh * S_ + s) * 256;
    v8f acc = (v8f){};
    for (int k0 = 0; k0 < 16; k0 += 4) {   // 16x16x16 product = 4x wmma_f32_16x16x4_f32
      const int kb = k0 + 2 * hlf;
      v2f a2, b2;
      const float m0e = dir ? Ms[kb * 16 + r16]       : Ms[r16 * 16 + kb];
      const float m1e = dir ? Ms[(kb + 1) * 16 + r16] : Ms[r16 * 16 + kb + 1];
      a2.x = aS * m0e + ((r16 == kb)     ? om : 0.f);
      a2.y = aS * m1e + ((r16 == kb + 1) ? om : 0.f);
      b2.x = T[cur][kb][r16];
      b2.y = T[cur][kb + 1][r16];
      acc = __builtin_amdgcn_wmma_f32_16x16x4_f32(false, a2, false, b2, (short)0, acc, false, false);
    }
    for (int r = 0; r < 8; ++r) T[cur ^ 1][r + 8 * hlf][r16] = acc[r];   // D layout
    __syncthreads();
    cur ^= 1;
  }
  float* dst = Tg + (size_t)((dir * BH_ + bh) * NCH_ + c) * 256;
  for (int e = lane; e < 256; e += 32) dst[e] = T[cur][e >> 4][e & 15];
}

// ---------------- K3b: short sequential scan over 128 chunk matrices per chain ----------------
__global__ __launch_bounds__(32) void k_scanU(
    const float* __restrict__ Tg, float* __restrict__ U)   // U: [2][BH_][NCH_+1][16]
{
  const int bh = blockIdx.x, dir = blockIdx.y;
  const int lane = threadIdx.x, r = lane & 15;
  float u[16];
  for (int j = 0; j < 16; ++j) u[j] = (j == 0) ? 1.f : 0.f;   // v0 = e0
  const float* Tb = Tg + (size_t)(dir * BH_ + bh) * NCH_ * 256;
  float* Ub = U + (size_t)(dir * BH_ + bh) * (NCH_ + 1) * 16;
  for (int cc = 0; cc < NCH_; ++cc) {
    if (lane < 16) Ub[cc * 16 + lane] = u[lane];
    const float* Tc = Tb + (size_t)cc * 256 + r * 16;
    float av = 0.f;
    for (int j = 0; j < 16; ++j) av += Tc[j] * u[j];
    for (int j = 0; j < 16; ++j) u[j] = __shfl(av, j, 32);
  }
  if (lane < 16) Ub[NCH_ * 16 + lane] = u[lane];   // final = v_global (dir 0)
}

// ---------------- K3c: refill per-step prefix vectors (8192-way parallel) ----------------
__global__ __launch_bounds__(32) void k_fillV(
    const float* __restrict__ Mn, const float* __restrict__ mask,
    const float* __restrict__ U, float* __restrict__ V)   // V: [2][BH_][S_][16]
{
  const int c = blockIdx.x, bh = blockIdx.y, dir = blockIdx.z;
  const int lane = threadIdx.x, r = lane & 15;
  const int b = bh >> 4;
  float v[16];
  const float* Uc = U + ((size_t)(dir * BH_ + bh) * (NCH_ + 1) + c) * 16;
  for (int j = 0; j < 16; ++j) v[j] = Uc[j];
  float* Vb = V + (size_t)(dir * BH_ + bh) * S_ * 16;
  for (int k = 0; k < CHK_; ++k) {
    const int i = c * CHK_ + k;
    const int s = dir ? (S_ - 1 - i) : i;
    if (lane < 16) Vb[(size_t)s * 16 + lane] = v[lane];    // pre-step value
    const float aS = mask[b * S_ + s], om = 1.f - aS;
    const float* Ms = Mn + ((size_t)bh * S_ + s) * 256;
    float av = 0.f;
    for (int j = 0; j < 16; ++j) {
      const float me = dir ? Ms[j * 16 + r] : Ms[r * 16 + j];
      av += (aS * me + ((j == r) ? om : 0.f)) * v[j];
    }
    for (int j = 0; j < 16; ++j) v[j] = __shfl(av, j, 32);
  }
}

// ---------------- K4: context gather + per-head 64x64 projection + GELU ----------------
__global__ __launch_bounds__(64) void k_ctx(
    const float* __restrict__ Mn, const float* __restrict__ mask,
    const float* __restrict__ U, const float* __restrict__ V,
    const float* __restrict__ Wv, const float* __restrict__ bv,
    float* __restrict__ out)
{
  const int bs = blockIdx.x, h = blockIdx.y;
  const int b = bs >> 11, s = bs & 2047;
  const int bh = b * 16 + h;
  const int t = threadIdx.x;
  __shared__ float x[64];
  if (t < 16) {                      // local: a*M[:,0] + (1-a)*e0
    const float aS = mask[b * S_ + s];
    x[t] = aS * Mn[((size_t)bh * S_ + s) * 256 + t * 16] + ((t == 0) ? (1.f - aS) : 0.f);
  } else if (t < 32) {               // global: lr prefix over all S
    x[t] = U[((size_t)(0 * BH_ + bh) * (NCH_ + 1) + NCH_) * 16 + (t - 16)];
  } else if (t < 48) {               // lr
    x[t] = V[((size_t)(0 * BH_ + bh) * S_ + s) * 16 + (t - 32)];
  } else {                           // rl
    x[t] = V[((size_t)(1 * BH_ + bh) * S_ + s) * 16 + (t - 48)];
  }
  __syncthreads();
  const float* Wh = Wv + (size_t)h * 64 * 64;
  float av = bv[h * 64 + t];
  for (int k = 0; k < 64; ++k) av += x[k] * Wh[k * 64 + t];
  const float g = 0.5f * av * (1.f + erff(av * 0.70710678118654752f));
  out[((size_t)(b * H_ + h) * S_ + s) * 64 + t] = g;   // (B,H,S,HV) flat == final view
}

extern "C" void kernel_launch(void* const* d_in, const int* in_sizes, int n_in,
                              void* d_out, int out_size, void* d_ws, size_t ws_size,
                              hipStream_t stream)
{
  (void)in_sizes; (void)n_in; (void)out_size; (void)ws_size;
  const float* hs   = (const float*)d_in[0];
  const float* mask = (const float*)d_in[1];
  const float* Wm   = (const float*)d_in[2];
  const float* bm   = (const float*)d_in[3];
  const float* Wv   = (const float*)d_in[4];
  const float* bv   = (const float*)d_in[5];
  float* out = (float*)d_out;

  char* ws = (char*)d_ws;
  size_t o = 0;
  unsigned short* hsb = (unsigned short*)(ws + o); o += (size_t)M_ * HID_ * 2;        // 16.8 MB
  unsigned short* wtb = (unsigned short*)(ws + o); o += (size_t)NCOL_ * HID_ * 2;     //  8.4 MB
  float* mbuf = (float*)(ws + o); o += (size_t)BH_ * S_ * 256 * 4;                    // 134 MB
  float* Tg   = (float*)(ws + o); o += (size_t)2 * BH_ * NCH_ * 256 * 4;              // 16.8 MB
  float* U    = (float*)(ws + o); o += (size_t)2 * BH_ * (NCH_ + 1) * 16 * 4;         //  1.1 MB
  float* V    = (float*)(ws + o);                                                     // 16.8 MB

  k_cvt_hs<<<(M_ * HID_ + 255) / 256, 256, 0, stream>>>(hs, hsb, M_ * HID_);
  k_cvt_wt<<<dim3(NCOL_ / 32, HID_ / 32), dim3(32, 8), 0, stream>>>(Wm, wtb);
  k_gemm<<<dim3(NCOL_ / TN, M_ / TM), 256, 0, stream>>>(hsb, wtb, bm, mbuf);
  k_norm<<<(BH_ * S_) / 8, 256, 0, stream>>>(mbuf);
  k_chunkT<<<dim3(NCH_, BH_, 2), 32, 0, stream>>>(mbuf, mask, Tg);
  k_scanU<<<dim3(BH_, 2), 32, 0, stream>>>(Tg, U);
  k_fillV<<<dim3(NCH_, BH_, 2), 32, 0, stream>>>(mbuf, mask, U, V);
  k_ctx<<<dim3(B_ * S_, H_), 64, 0, stream>>>(mbuf, mask, U, V, Wv, bv, out);
}
